// DinoText_90323162235086
// MI455X (gfx1250) — compile-verified, matrix operand order checked by reading, not run on hardware
//
#include <hip/hip_runtime.h>
#include <hip/hip_bf16.h>
#include <math.h>

typedef __attribute__((ext_vector_type(16))) _Float16 v16h;
typedef __attribute__((ext_vector_type(8)))  float    v8f;

#define B_DIM   1024
#define P_DIM   256
#define D_DINO  1024
#define D_CLIP  512

// Fast tanh: 1 - 2/(exp(2x)+1). Saturates correctly (exp->inf => 1, exp->0 => -1),
// monotone; avoids libm tanhf's branchy exec-divergent expansion.
__device__ __forceinline__ float fast_tanh(float x) {
    float e = __expf(2.0f * x);
    return 1.0f - 2.0f / (e + 1.0f);
}

// Load a 16-half WMMA fragment for one lane from a row-major row (K contiguous).
// Lane layout (ISA 7.12.2, 16-bit A 16x32): halves are 8 contiguous K values at
// kk + hl*8 and kk + 16 + hl*8  -> two pairs of aligned float4 loads.
__device__ __forceinline__ v16h load_frag(const float* __restrict__ row, int kk, int hl) {
    const float4* p0 = (const float4*)(row + kk + hl * 8);
    const float4* p1 = (const float4*)(row + kk + 16 + hl * 8);
    const float4 x0 = p0[0], x1 = p0[1];
    const float4 x2 = p1[0], x3 = p1[1];
    v16h f;
    f[0]  = (_Float16)x0.x; f[1]  = (_Float16)x0.y; f[2]  = (_Float16)x0.z; f[3]  = (_Float16)x0.w;
    f[4]  = (_Float16)x1.x; f[5]  = (_Float16)x1.y; f[6]  = (_Float16)x1.z; f[7]  = (_Float16)x1.w;
    f[8]  = (_Float16)x2.x; f[9]  = (_Float16)x2.y; f[10] = (_Float16)x2.z; f[11] = (_Float16)x2.w;
    f[12] = (_Float16)x3.x; f[13] = (_Float16)x3.y; f[14] = (_Float16)x3.z; f[15] = (_Float16)x3.w;
    return f;
}

// ---------------------------------------------------------------------------
// Kernel 1: t_raw[B, D_DINO] = tanh(text[B, D_CLIP] @ W[D_DINO, D_CLIP]^T + b)
// One 16x16 tile per wave, f16 WMMA, f32 accumulate.
// ---------------------------------------------------------------------------
__global__ void proj_wmma(const float* __restrict__ text,
                          const float* __restrict__ W,
                          const float* __restrict__ bias,
                          float* __restrict__ t_out) {
    const int tid  = threadIdx.x;
    const int wave = tid >> 5;
    const int lane = tid & 31;
    const int tile = blockIdx.x * 8 + wave;     // 0..4095
    const int tm   = (tile >> 6) << 4;          // batch-row block
    const int tn   = (tile & 63) << 4;          // dino-col block
    const int hl   = lane >> 4;
    const int r    = lane & 15;

    v8f acc = {};
    const float* arow = text + (size_t)(tm + r) * D_CLIP;
    const float* brow = W    + (size_t)(tn + r) * D_CLIP;   // B[k][n] = W[n][k]

#pragma unroll 2
    for (int kk = 0; kk < D_CLIP; kk += 32) {
        v16h a  = load_frag(arow, kk, hl);
        v16h bf = load_frag(brow, kk, hl);
        acc = __builtin_amdgcn_wmma_f32_16x16x32_f16(
                  false, a, false, bf, (short)0, acc, false, false);
    }

    const int col = tn + r;
    const float bb = bias[col];
#pragma unroll
    for (int q = 0; q < 8; ++q) {
        const int row = tm + hl * 8 + q;
        t_out[(size_t)row * D_DINO + col] = fast_tanh(acc[q] + bb);
    }
}

// ---------------------------------------------------------------------------
// Kernel 2: in-place L2 row-normalize t  (x / max(||x||, 1e-12))
// ---------------------------------------------------------------------------
__global__ void rownorm(float* __restrict__ t) {
    const int row = blockIdx.x;
    const int tid = threadIdx.x;
    float4* p = (float4*)(t + (size_t)row * D_DINO);
    float4 v = p[tid];
    float s = v.x*v.x + v.y*v.y + v.z*v.z + v.w*v.w;
#pragma unroll
    for (int o = 16; o > 0; o >>= 1) s += __shfl_xor(s, o, 32);
    __shared__ float sm[8];
    if ((tid & 31) == 0) sm[tid >> 5] = s;
    __syncthreads();
    float tot = 0.f;
#pragma unroll
    for (int w = 0; w < 8; ++w) tot += sm[w];
    const float inv = 1.0f / fmaxf(sqrtf(tot), 1e-12f);
    v.x *= inv; v.y *= inv; v.z *= inv; v.w *= inv;
    p[tid] = v;
}

// ---------------------------------------------------------------------------
// Kernel 3: per batch b, single pass over visual[b] (1 MiB):
//   sims[j] = dot(t_n[b], v[b,j]) / max(||v[b,j]||, eps); argmax over j.
//   (softmax skipped: monotone, argmax-preserving)
// Wave w owns patches [w*32, w*32+32): each lane keeps 32 columns of t[b] in
// registers, reductions are pure wave32 shuffles -> no barriers in the loop.
// One LDS arg-reduce of 8 per-wave candidates at the end (ascending wave scan
// preserves first-occurrence argmax semantics).
// ---------------------------------------------------------------------------
__global__ void best_patch(const float* __restrict__ visual,
                           const float* __restrict__ t_n,
                           float* __restrict__ v_best) {
    const int b    = blockIdx.x;
    const int tid  = threadIdx.x;            // 256
    const int wave = tid >> 5;
    const int lane = tid & 31;

    const float4* vb4 = (const float4*)(visual + (size_t)b * P_DIM * D_DINO);
    const float4* t4  = (const float4*)(t_n + (size_t)b * D_DINO);

    float4 treg[8];
#pragma unroll
    for (int c = 0; c < 8; ++c) treg[c] = t4[c * 32 + lane];

    float best_sim = -3.402823e38f;
    int   best_j   = 0;
    float best_inv = 0.f;

    for (int jj = 0; jj < 32; ++jj) {
        const int j = wave * 32 + jj;
        const float4* row = vb4 + (size_t)j * (D_DINO / 4);
        if (jj < 31) {
            // next patch row: lanes striped 128B apart cover the whole 4KiB row
            __builtin_prefetch(row + (D_DINO / 4) + lane * 8, 0, 0);
        }
        float d = 0.f, n = 0.f;
#pragma unroll
        for (int c = 0; c < 8; ++c) {
            const float4 vv = row[c * 32 + lane];
            const float4 tt = treg[c];
            d += vv.x*tt.x + vv.y*tt.y + vv.z*tt.z + vv.w*tt.w;
            n += vv.x*vv.x + vv.y*vv.y + vv.z*vv.z + vv.w*vv.w;
        }
#pragma unroll
        for (int o = 16; o > 0; o >>= 1) {
            d += __shfl_xor(d, o, 32);
            n += __shfl_xor(n, o, 32);
        }
        const float inv = 1.0f / fmaxf(sqrtf(n), 1e-12f);
        const float sim = d * inv;
        if (sim > best_sim) { best_sim = sim; best_j = j; best_inv = inv; }
    }

    __shared__ float s_sim[8];
    __shared__ int   s_j[8];
    __shared__ float s_inv[8];
    if (lane == 0) { s_sim[wave] = best_sim; s_j[wave] = best_j; s_inv[wave] = best_inv; }
    __syncthreads();

    float bs = s_sim[0]; int bj = s_j[0]; float bi = s_inv[0];
#pragma unroll
    for (int w = 1; w < 8; ++w) {
        if (s_sim[w] > bs) { bs = s_sim[w]; bj = s_j[w]; bi = s_inv[w]; }
    }

    const float4 bv = vb4[(size_t)bj * (D_DINO / 4) + tid];
    float4 o;
    o.x = bv.x * bi; o.y = bv.y * bi; o.z = bv.z * bi; o.w = bv.w * bi;
    ((float4*)(v_best + (size_t)b * D_DINO))[tid] = o;
}

// ---------------------------------------------------------------------------
// Kernel 4: out[B, B] = t_n @ v_best^T  (K = 1024), WMMA.
// ---------------------------------------------------------------------------
__global__ void out_wmma(const float* __restrict__ t_n,
                         const float* __restrict__ v_best,
                         float* __restrict__ out) {
    const int tid  = threadIdx.x;
    const int wave = tid >> 5;
    const int lane = tid & 31;
    const int tile = blockIdx.x * 8 + wave;
    const int tm   = (tile >> 6) << 4;
    const int tn   = (tile & 63) << 4;
    const int hl   = lane >> 4;
    const int r    = lane & 15;

    v8f acc = {};
    const float* arow = t_n    + (size_t)(tm + r) * D_DINO;
    const float* brow = v_best + (size_t)(tn + r) * D_DINO;

#pragma unroll 2
    for (int kk = 0; kk < D_DINO; kk += 32) {
        v16h a  = load_frag(arow, kk, hl);
        v16h bf = load_frag(brow, kk, hl);
        acc = __builtin_amdgcn_wmma_f32_16x16x32_f16(
                  false, a, false, bf, (short)0, acc, false, false);
    }

    const int col = tn + r;
#pragma unroll
    for (int q = 0; q < 8; ++q) {
        const int row = tm + hl * 8 + q;
        out[(size_t)row * B_DIM + col] = acc[q];
    }
}

// ---------------------------------------------------------------------------
extern "C" void kernel_launch(void* const* d_in, const int* in_sizes, int n_in,
                              void* d_out, int out_size, void* d_ws, size_t ws_size,
                              hipStream_t stream) {
    const float* visual = (const float*)d_in[0];  // [1024, 256, 1024]
    const float* text   = (const float*)d_in[1];  // [1024, 512]
    const float* W      = (const float*)d_in[2];  // [1024, 512]
    const float* bias   = (const float*)d_in[3];  // [1024]
    float* out = (float*)d_out;                   // [1024, 1024]

    float* t_ws  = (float*)d_ws;                                  // 4 MiB
    float* vb_ws = (float*)((char*)d_ws + (size_t)B_DIM * D_DINO * sizeof(float));

    const int tiles  = (B_DIM / 16) * (D_DINO / 16);  // 4096
    const int blocks = tiles / 8;                     // 512

    proj_wmma<<<dim3(blocks), dim3(256), 0, stream>>>(text, W, bias, t_ws);
    rownorm<<<dim3(B_DIM), dim3(256), 0, stream>>>(t_ws);
    best_patch<<<dim3(B_DIM), dim3(256), 0, stream>>>(visual, t_ws, vb_ws);
    out_wmma<<<dim3(blocks), dim3(256), 0, stream>>>(t_ws, vb_ws, out);
}
